// DPINeuron_73770358276702
// MI455X (gfx1250) — matrix operand order repaired and checked by reading, not compiled, not run on hardware
//
#include <hip/hip_runtime.h>
#include <hip/hip_bf16.h>
#include <math.h>

// ---------------------------------------------------------------------------
// DPI neuron step, MI455X (gfx1250, wave32).
//   Phase 1: fp32 -> fp16 pack (X as-is; W with round-to-nearest-even = STE round fwd)
//   Phase 2: dual GEMM on v_wmma_f32_16x16x32_f16 + fused neuron-state epilogue.
// numSyn counts are integers <= 2048 -> exact in f16 inputs / f32 accumulation.
//
// Tiling: wave tile = 64(M) x 32(N) = 4x2 WMMA tiles for BOTH weight matrices.
//   Per K=32 step: 8 A b128-loads + 8 B b128-loads feed 16 WMMAs (1.0 ld/wmma).
// Workgroup = 8 waves (2Mx4N) -> 128x128 macro-tile. Grid = (4096/128, 2048/128).
// GEMM operands are L2-resident (32 MB fp16 vs 192 MB L2); single-use state
// traffic uses non-temporal hints so it doesn't evict them. sched_barriers
// fence the epilogue per-tile so accumulators retire without scratch spills.
// ---------------------------------------------------------------------------

typedef __attribute__((ext_vector_type(16))) _Float16 v16h;
typedef __attribute__((ext_vector_type(8)))  _Float16 v8h;
typedef __attribute__((ext_vector_type(4)))  _Float16 v4h;
typedef __attribute__((ext_vector_type(8)))  float    v8f;

#define B_DIM 4096
#define K_DIM 2048
#define N_DIM 2048

// ---- physical constants (match reference) ----
#define C_I0        5e-13f
#define C_DT        1e-3f
#define C_TAU_MEM   0.10638297872340426f   // UT/KAPPA * CMEM/ITAU_MEM
#define C_TAU_AMPA  0.07092198581560284f   // UT/KAPPA * CAMPA/ITAU_AMPA
#define C_TAU_SHUNT 0.07092198581560284f   // UT/KAPPA * CSHUNT/ITAU_AMPA
#define C_ITAU_MEM  1e-12f
#define C_IGAIN_MEM 1e-12f
#define C_ITH       1e-12f
#define C_IPFB_TH   1e-12f
#define C_IPFB_NORM 1e-12f
#define C_POW_EXP   0.41348973607038125f   // KAPPA/(KAPPA+1)
#define C_I0_POW    6.0992e-8f             // I0^(1/(KAPPA+1))

// ---------------------------------------------------------------------------
// fp32 -> fp16 conversion (vectorized x4), optional round-to-nearest-even.
// ---------------------------------------------------------------------------
__global__ void dpi_pack_f16(const float* __restrict__ src,
                             _Float16* __restrict__ dst,
                             int n4, int do_round) {
  int i = blockIdx.x * blockDim.x + threadIdx.x;
  if (i >= n4) return;
  float4 v = reinterpret_cast<const float4*>(src)[i];
  if (do_round) {
    v.x = rintf(v.x); v.y = rintf(v.y); v.z = rintf(v.z); v.w = rintf(v.w);
  }
  v4h o;
  o[0] = (_Float16)v.x; o[1] = (_Float16)v.y;
  o[2] = (_Float16)v.z; o[3] = (_Float16)v.w;
  reinterpret_cast<v4h*>(dst)[i] = o;
}

// ---------------------------------------------------------------------------
// Fused neuron state update for one C-matrix element.
// ---------------------------------------------------------------------------
__device__ __forceinline__ void dpi_neuron(
    float numA, float numS,
    float imem, float iampa, float ishunt, float refr,
    float Idc, float IwA, float IwS, float alpha, float beta,
    float& spk, float& imem_o, float& iampa_o, float& ishunt_o, float& refr_o)
{
  float dIampa  = -iampa  / C_TAU_AMPA;
  float iampa1  = iampa  + IwA * numA;          // IGAIN_AMPA/ITAU_AMPA == 1
  float dIshunt = -ishunt / C_TAU_SHUNT;
  float ishunt1 = ishunt + IwS * numS;

  float Iin = Idc + iampa1 + C_I0 - ishunt1;    // Inmda = I0
  Iin = (refr <= 0.0f) ? Iin : 0.0f;
  Iin = fmaxf(Iin, C_I0);

  float Ifb = C_I0_POW * __powf(imem, C_POW_EXP)
              / (1.0f + __expf(-C_IPFB_NORM * (imem - C_IPFB_TH)));
  float f_imem = Ifb / C_ITAU_MEM * (imem + C_IGAIN_MEM);
  float dImem = (alpha * (Iin - C_ITAU_MEM - C_I0) - beta * imem + f_imem)
                / (C_TAU_MEM * (1.0f + C_IGAIN_MEM / imem));
  float imem1 = fmaxf(imem + dImem * C_DT, C_I0);

  // faithful to reference: dIshunt decay also applied to Iampa
  float iampa2 = fmaxf(iampa1 + dIampa  * C_DT, C_I0);
  iampa2       = fmaxf(iampa2 + dIshunt * C_DT, C_I0);

  spk = (imem1 - C_ITH > 0.0f) ? 1.0f : 0.0f;
  imem_o   = (1.0f - spk) * imem1 + spk * C_I0;
  float r1 = fmaxf(refr - C_DT, 0.0f);
  refr_o   = (1.0f - spk) * r1;                 // REFP = 0
  iampa_o  = iampa2;
  ishunt_o = ishunt1;
}

// ---------------------------------------------------------------------------
// Dual-GEMM + fused epilogue.
// ---------------------------------------------------------------------------
__global__ __launch_bounds__(256)
void dpi_gemm_neuron(const _Float16* __restrict__ Xh,
                     const _Float16* __restrict__ Wa,
                     const _Float16* __restrict__ Ws,
                     const float* __restrict__ Imem_in,
                     const float* __restrict__ Iampa_in,
                     const float* __restrict__ Ishunt_in,
                     const float* __restrict__ Refr_in,
                     const float* __restrict__ sIdc,
                     const float* __restrict__ sIwA,
                     const float* __restrict__ sIwS,
                     const float* __restrict__ sAlpha,
                     const float* __restrict__ sBeta,
                     float* __restrict__ out)
{
  const int lane = threadIdx.x & 31;
  const int wid  = threadIdx.x >> 5;
  const int wm   = wid & 1;                    // wave M index (0..1)
  const int wn   = wid >> 1;                   // wave N index (0..3)
  const int bm0  = blockIdx.x * 128 + wm * 64; // output row base for this wave
  const int bn0  = blockIdx.y * 128 + wn * 32; // output col base for this wave

  const int lrow = lane & 15;                  // lane's row/col within 16
  const int lhi  = lane >> 4;                  // lane half (K sub-block select)

  // A-fragment rows (X, row-major, K contiguous): 4 M-tiles
  const _Float16* xr[4];
#pragma unroll
  for (int mi = 0; mi < 4; ++mi)
    xr[mi] = Xh + (size_t)(bm0 + mi * 16 + lrow) * K_DIM;

  // B-fragment "columns" = W rows (row-major, K contiguous): 2 N-tiles, 2 mats
  const _Float16* war[2];
  const _Float16* wsr[2];
#pragma unroll
  for (int ni = 0; ni < 2; ++ni) {
    war[ni] = Wa + (size_t)(bn0 + ni * 16 + lrow) * K_DIM;
    wsr[ni] = Ws + (size_t)(bn0 + ni * 16 + lrow) * K_DIM;
  }

  const int ka_off = lhi * 16;  // A: lanes 16-31 hold K+16..K+31
  const int kb_off = lhi * 16;  // B: lanes 16-31 hold K+16..K+31

  v8f accA[4][2], accS[4][2];
#pragma unroll
  for (int mi = 0; mi < 4; ++mi)
#pragma unroll
    for (int ni = 0; ni < 2; ++ni) { accA[mi][ni] = 0; accS[mi][ni] = 0; }

#pragma unroll 1
  for (int k0 = 0; k0 < K_DIM; k0 += 32) {
    // issue all fragment loads first, then consume with partial waits
    v16h ba[2], bs[2];
#pragma unroll
    for (int ni = 0; ni < 2; ++ni) {
      ba[ni] = *reinterpret_cast<const v16h*>(war[ni] + k0 + kb_off);
      bs[ni] = *reinterpret_cast<const v16h*>(wsr[ni] + k0 + kb_off);
    }
    union { v16h v; v8h h[2]; } a[4];
#pragma unroll
    for (int mi = 0; mi < 4; ++mi) {
      a[mi].h[0] = *reinterpret_cast<const v8h*>(xr[mi] + k0 + ka_off);
      a[mi].h[1] = *reinterpret_cast<const v8h*>(xr[mi] + k0 + ka_off + 8);
    }

    // speculative prefetch ~4 cachelines ahead, folded onto the load address
    // registers (silently dropped past the buffer end; no guard branch needed)
    __builtin_prefetch(xr[0]   + k0 + ka_off + 256, 0, 3);
    __builtin_prefetch(war[0]  + k0 + kb_off + 256, 0, 3);
    __builtin_prefetch(wsr[0]  + k0 + kb_off + 256, 0, 3);

#pragma unroll
    for (int mi = 0; mi < 4; ++mi)
#pragma unroll
      for (int ni = 0; ni < 2; ++ni) {
        accA[mi][ni] = __builtin_amdgcn_wmma_f32_16x16x32_f16(
            false, a[mi].v, false, ba[ni], (short)0, accA[mi][ni], false, false);
        accS[mi][ni] = __builtin_amdgcn_wmma_f32_16x16x32_f16(
            false, a[mi].v, false, bs[ni], (short)0, accS[mi][ni], false, false);
      }
  }

  // ---- fused epilogue ----
  const float Idc = sIdc[0], IwA = sIwA[0], IwS = sIwS[0];
  const float alpha = sAlpha[0], beta = sBeta[0];

  const size_t PLANE = (size_t)B_DIM * N_DIM;
  float* out_spike  = out;
  float* out_imem   = out + PLANE;
  float* out_iampa  = out + 2 * PLANE;
  float* out_ishunt = out + 3 * PLANE;
  float* out_refr   = out + 4 * PLANE;

#pragma unroll
  for (int mi = 0; mi < 4; ++mi)
#pragma unroll
    for (int ni = 0; ni < 2; ++ni) {
      const int col = bn0 + ni * 16 + lrow;   // C/D: N = lane & 15
#pragma unroll
      for (int r = 0; r < 8; ++r) {
        // C/D layout: VGPR r -> M = r (lanes 0-15) / r+8 (lanes 16-31)
        const int row = bm0 + mi * 16 + r + 8 * lhi;
        const size_t idx = (size_t)row * N_DIM + col;

        // single-use state traffic: non-temporal so GEMM operands stay cached
        float imem   = __builtin_nontemporal_load(Imem_in   + idx);
        float iampa  = __builtin_nontemporal_load(Iampa_in  + idx);
        float ishunt = __builtin_nontemporal_load(Ishunt_in + idx);
        float refr   = __builtin_nontemporal_load(Refr_in   + idx);

        float spk, imem_o, iampa_o, ishunt_o, refr_o;
        dpi_neuron(accA[mi][ni][r], accS[mi][ni][r],
                   imem, iampa, ishunt, refr,
                   Idc, IwA, IwS, alpha, beta,
                   spk, imem_o, iampa_o, ishunt_o, refr_o);

        __builtin_nontemporal_store(spk,      out_spike  + idx);
        __builtin_nontemporal_store(imem_o,   out_imem   + idx);
        __builtin_nontemporal_store(iampa_o,  out_iampa  + idx);
        __builtin_nontemporal_store(ishunt_o, out_ishunt + idx);
        __builtin_nontemporal_store(refr_o,   out_refr   + idx);
      }
      // fence: retire this tile's accumulators/loads before starting the next
      // tile, keeping peak VGPR pressure under the 256-reg window (no spills)
      __builtin_amdgcn_sched_barrier(0);
    }
}

// ---------------------------------------------------------------------------
extern "C" void kernel_launch(void* const* d_in, const int* in_sizes, int n_in,
                              void* d_out, int out_size, void* d_ws, size_t ws_size,
                              hipStream_t stream) {
  const float* X       = (const float*)d_in[0];
  const float* W_ampa  = (const float*)d_in[1];
  const float* W_shunt = (const float*)d_in[2];
  const float* Imem    = (const float*)d_in[3];
  const float* Iampa   = (const float*)d_in[4];
  const float* Ishunt  = (const float*)d_in[5];
  const float* Refr    = (const float*)d_in[6];
  const float* Idc     = (const float*)d_in[7];
  const float* IwA     = (const float*)d_in[8];
  const float* IwS     = (const float*)d_in[9];
  const float* alpha   = (const float*)d_in[10];
  const float* beta    = (const float*)d_in[11];

  // scratch layout: Xh (16MB) | Wa_h (8MB) | Ws_h (8MB)
  _Float16* Xh  = (_Float16*)d_ws;
  _Float16* Wah = Xh  + (size_t)B_DIM * K_DIM;
  _Float16* Wsh = Wah + (size_t)N_DIM * K_DIM;

  {
    int n4 = B_DIM * K_DIM / 4;
    dpi_pack_f16<<<(n4 + 255) / 256, 256, 0, stream>>>(X, Xh, n4, 0);
    n4 = N_DIM * K_DIM / 4;
    dpi_pack_f16<<<(n4 + 255) / 256, 256, 0, stream>>>(W_ampa,  Wah, n4, 1);
    dpi_pack_f16<<<(n4 + 255) / 256, 256, 0, stream>>>(W_shunt, Wsh, n4, 1);
  }

  dim3 grid(B_DIM / 128, N_DIM / 128);
  dpi_gemm_neuron<<<grid, 256, 0, stream>>>(
      Xh, Wah, Wsh, Imem, Iampa, Ishunt, Refr,
      Idc, IwA, IwS, alpha, beta, (float*)d_out);
}
